// ASD_72069551227061
// MI455X (gfx1250) — matrix operand order repaired and checked by reading, not compiled, not run on hardware
//
#include <hip/hip_runtime.h>
#include <hip/hip_bf16.h>

// ---------------------------------------------------------------------------
// MI455X (gfx1250) implementation.
//  - GCN edge aggregation: thread-per-edge float atomics (L2-resident node
//    arrays, streamed edge list with global_prefetch_b8).  ~18 GB of edge
//    traffic dominates the whole pipeline (~0.8 ms @ 23.3 TB/s).
//  - Dense GEMMs (node transform h@W and fc_in 128x9800x4340) use
//    v_wmma_f32_16x16x32_f16 (f16 A/B, f32 accumulate). xform stages tiles in
//    zero-padded f16 LDS with 16B-aligned contiguous fragment reads
//    (ds_load_b128); fc1 uses branch-free float4 global loads. Store
//    epilogues are unguarded on full tiles (no per-element exec juggling).
//  - BiLSTM: block per (dir,batch-row), weights in LDS.
// ---------------------------------------------------------------------------

typedef __attribute__((ext_vector_type(16))) _Float16 v16h;
typedef __attribute__((ext_vector_type(8)))  _Float16 v8h;
typedef __attribute__((ext_vector_type(8)))  float    v8f;

#define DD   35
#define BB   128
#define DHH  70
#define TTT  140

__device__ __forceinline__ v8f wmma16(v16h a, v16h b, v8f c) {
  // (neg_a, A, neg_b, B, c_mod, C, reuse_a, reuse_b)
  return __builtin_amdgcn_wmma_f32_16x16x32_f16(false, a, false, b, (short)0, c,
                                                false, false);
}

__device__ __forceinline__ v16h cat8(v8h lo, v8h hi) {
  return __builtin_shufflevector(lo, hi, 0, 1, 2, 3, 4, 5, 6, 7,
                                 8, 9, 10, 11, 12, 13, 14, 15);
}

__device__ __forceinline__ float sigmf(float x) { return 1.0f / (1.0f + __expf(-x)); }

__device__ __forceinline__ void pack8(v16h& dst, int off, float4 lo, float4 hi) {
  dst[off + 0] = (_Float16)lo.x; dst[off + 1] = (_Float16)lo.y;
  dst[off + 2] = (_Float16)lo.z; dst[off + 3] = (_Float16)lo.w;
  dst[off + 4] = (_Float16)hi.x; dst[off + 5] = (_Float16)hi.y;
  dst[off + 6] = (_Float16)hi.z; dst[off + 7] = (_Float16)hi.w;
}

// ---------------------------------------------------------------- fill zero
__global__ void k_fill0(float* __restrict__ p, size_t n) {
  size_t i  = (size_t)blockIdx.x * blockDim.x + threadIdx.x;
  size_t st = (size_t)gridDim.x * blockDim.x;
  for (; i < n; i += st) p[i] = 0.0f;
}

// ---------------------------------------------------------------- degrees
__global__ void k_deg(const int* __restrict__ src, const int* __restrict__ dst,
                      const int* __restrict__ split,
                      float* __restrict__ degp, float* __restrict__ degl, int E) {
  int e = blockIdx.x * blockDim.x + threadIdx.x;
  if (e >= E) return;
  __builtin_prefetch(&src[e + 8192], 0, 1);
  __builtin_prefetch(&dst[e + 8192], 0, 1);
  int s = src[e], d = dst[e];
  if (split[s] == 1) atomicAdd(&degp[d], 1.0f);
  else               atomicAdd(&degl[d], 1.0f);
}

__global__ void k_dinv(float* __restrict__ degp, float* __restrict__ degl, int N) {
  int i = blockIdx.x * blockDim.x + threadIdx.x;
  if (i >= N) return;
  degp[i] = rsqrtf(degp[i] + 1.0f);   // in-place deg -> dinv
  degl[i] = rsqrtf(degl[i] + 1.0f);
}

// ------------------------------------------------- node transform t = h @ W
// Block = 256 threads (8 waves) = 128 nodes. Tiles staged in zero-padded f16
// LDS; W stored transposed so every fragment read is two contiguous,
// 16B-aligned ds_load_b128s. Full tiles take an unguarded store path.
__global__ void k_xform_wmma(const float* __restrict__ h,
                             const float* __restrict__ Wt,   // [35,35] (in,out)
                             float* __restrict__ t, int nnodes) {
  __shared__ _Float16 s_h[128][48];   // [node][K]  (zero padded)
  __shared__ _Float16 s_wt[48][48];   // [N][K] transposed (zero padded)

  int tid  = threadIdx.x;
  int lane = tid & 31;
  int wid  = tid >> 5;
  int base = blockIdx.x * 128;
  bool fulltile = (base + 128 <= nnodes);   // block-uniform

  // zero everything (covers the pad), then load the valid region
  for (int i = tid; i < 128 * 48; i += 256) (&s_h[0][0])[i] = (_Float16)0.0f;
  for (int i = tid; i < 48 * 48;  i += 256) (&s_wt[0][0])[i] = (_Float16)0.0f;
  __syncthreads();
  if (fulltile) {
    const float* hb = h + (size_t)base * DD;   // tile is one contiguous range
    for (int i = tid; i < 128 * DD; i += 256) {
      int r = i / DD, c = i - r * DD;
      s_h[r][c] = (_Float16)hb[i];
    }
  } else {
    for (int i = tid; i < 128 * DD; i += 256) {
      int r = i / DD, c = i - r * DD;
      int node = base + r;
      float v = (node < nnodes) ? h[(size_t)node * DD + c] : 0.0f;
      s_h[r][c] = (_Float16)v;
    }
  }
  for (int i = tid; i < DD * DD; i += 256) {
    int kx = i / DD, nx = i - kx * DD;
    s_wt[nx][kx] = (_Float16)Wt[i];     // transpose into [N][K]
  }
  __syncthreads();

  int m0 = base + wid * 16;
  if (m0 < nnodes) {                       // wave-uniform
    int mr = wid * 16 + (lane & 15);
    int ka = (lane >> 4) * 8;
    int kb = (lane >> 4) * 16;
    int nl = lane & 15;

    v8f acc[3];
#pragma unroll
    for (int c = 0; c < 3; ++c) acc[c] = {};

#pragma unroll
    for (int ks = 0; ks < 2; ++ks) {
      int k0 = ks * 32;
      // A fragment: two contiguous 8-half runs (16B-aligned)
      v8h alo = *(const v8h*)&s_h[mr][k0 + ka];
      v8h ahi = *(const v8h*)&s_h[mr][k0 + 16 + ka];
      v16h a  = cat8(alo, ahi);
#pragma unroll
      for (int ct = 0; ct < 3; ++ct) {
        // B fragment: two contiguous 8-half runs of one W^T row (same k order as A)
        v8h blo = *(const v8h*)&s_wt[ct * 16 + nl][k0 + ka];
        v8h bhi = *(const v8h*)&s_wt[ct * 16 + nl][k0 + 16 + ka];
        acc[ct] = wmma16(a, cat8(blo, bhi), acc[ct]);
      }
    }
    // C layout: VGPR g, lane l -> (M = m0 + g + 8*(l>>4), N = ct*16 + (l&15))
    int mb = m0 + 8 * (lane >> 4);
    if (fulltile) {
#pragma unroll
      for (int ct = 0; ct < 3; ++ct) {
        int n = ct * 16 + nl;                 // ct<2: provably < 35
        if (n < DD) {
#pragma unroll
          for (int g = 0; g < 8; ++g)
            t[(size_t)(mb + g) * DD + n] = acc[ct][g];
        }
      }
    } else {
#pragma unroll
      for (int ct = 0; ct < 3; ++ct) {
        int n = ct * 16 + nl;
        if (n < DD) {
#pragma unroll
          for (int g = 0; g < 8; ++g) {
            int m = mb + g;
            if (m < nnodes) t[(size_t)m * DD + n] = acc[ct][g];
          }
        }
      }
    }
  }
}

// ---------------------------------------------------------- edge aggregation
__global__ void k_edge_agg(const int* __restrict__ src, const int* __restrict__ dst,
                           const int* __restrict__ split, int want,
                           const float* __restrict__ dinv,
                           const float* __restrict__ t,
                           float* __restrict__ agg, int E) {
  int e = blockIdx.x * blockDim.x + threadIdx.x;
  if (e >= E) return;
  __builtin_prefetch(&src[e + 8192], 0, 1);
  __builtin_prefetch(&dst[e + 8192], 0, 1);
  int s = src[e];
  if (split[s] != want) return;
  int d = dst[e];
  float w = dinv[s] * dinv[d];
  const float* ts = t + (size_t)s * DD;
  float*       ad = agg + (size_t)d * DD;
#pragma unroll
  for (int j = 0; j < DD; ++j) atomicAdd(&ad[j], ts[j] * w);
}

// ------------------------------------------------ h = relu(agg + t*dinv^2 + b)
__global__ void k_node_update(const float* __restrict__ t, const float* __restrict__ agg,
                              const float* __restrict__ dinv, const float* __restrict__ bias,
                              float* __restrict__ hout, int total) {
  int i = blockIdx.x * blockDim.x + threadIdx.x;
  if (i >= total) return;
  int node = i / DD, j = i - node * DD;
  float dv = dinv[node];
  float v  = agg[i] + t[i] * dv * dv + bias[j];
  hout[i]  = fmaxf(v, 0.0f);
}

// --------------------------------------------------------------- graph pool
__global__ void k_pool(const float* __restrict__ h, const int* __restrict__ batch,
                       float* __restrict__ pooled, int total) {
  int i = blockIdx.x * blockDim.x + threadIdx.x;
  if (i >= total) return;
  int node = i / DD, j = i - node * DD;
  atomicAdd(&pooled[batch[node] * DD + j], h[i]);
}

// ------------------------------------------------------------------- BiLSTM
// grid = 256 blocks: blockIdx<128 = forward, else backward. block=160 threads.
__global__ void k_lstm(const float* __restrict__ lig, const float* __restrict__ pro,
                       const float* __restrict__ wif, const float* __restrict__ whf,
                       const float* __restrict__ bif, const float* __restrict__ bhf,
                       const float* __restrict__ wib, const float* __restrict__ whb,
                       const float* __restrict__ bib, const float* __restrict__ bhb,
                       float* __restrict__ out /*[B,140,70]*/) {
  int b   = blockIdx.x & (BB - 1);
  int dir = blockIdx.x >> 7;
  const float* wih = dir ? wib : wif;
  const float* whh = dir ? whb : whf;
  const float* bi  = dir ? bib : bif;
  const float* bh  = dir ? bhb : bhf;

  __shared__ float s_wih[4 * DD * DD];   // [140,35]
  __shared__ float s_whh[4 * DD * DD];
  __shared__ float s_b[4 * DD];
  __shared__ float s_x[2][DD];
  __shared__ float s_h[DD], s_c[DD], s_g[4 * DD];

  int tid = threadIdx.x;
  for (int i = tid; i < 4 * DD * DD; i += blockDim.x) { s_wih[i] = wih[i]; s_whh[i] = whh[i]; }
  for (int i = tid; i < 4 * DD; i += blockDim.x) s_b[i] = bi[i] + bh[i];
  if (tid < DD) {
    s_x[0][tid] = lig[b * DD + tid];
    s_x[1][tid] = pro[b * DD + tid];
    s_h[tid] = 0.0f; s_c[tid] = 0.0f;
  }
  __syncthreads();

  for (int s = 0; s < TTT; ++s) {
    int tt = dir ? (TTT - 1 - s) : s;
    if (tid < 4 * DD) {
      float g = s_b[tid];
      const float* wr = &s_whh[tid * DD];
      if (tt < 2) {
        const float* x  = s_x[tt];
        const float* wi = &s_wih[tid * DD];
#pragma unroll
        for (int j = 0; j < DD; ++j) g += x[j] * wi[j];
      }
#pragma unroll
      for (int j = 0; j < DD; ++j) g += s_h[j] * wr[j];
      s_g[tid] = g;
    }
    __syncthreads();
    if (tid < DD) {
      float gi = s_g[tid], gf = s_g[DD + tid], gg = s_g[2 * DD + tid], go = s_g[3 * DD + tid];
      float c  = sigmf(gf) * s_c[tid] + sigmf(gi) * tanhf(gg);
      float h  = sigmf(go) * tanhf(c);
      s_c[tid] = c; s_h[tid] = h;
      out[((size_t)b * TTT + tt) * DHH + dir * DD + tid] = h;
    }
    __syncthreads();
  }
}

// ----------------------------------------------------------- q/k/v projection
__global__ void k_qkv(const float* __restrict__ seq,
                      const float* __restrict__ Wq, const float* __restrict__ bq,
                      const float* __restrict__ Wk, const float* __restrict__ bk,
                      const float* __restrict__ Wv, const float* __restrict__ bv,
                      float* __restrict__ q, float* __restrict__ k, float* __restrict__ v) {
  const int TOT = BB * TTT * DHH;
  int i = blockIdx.x * blockDim.x + threadIdx.x;
  if (i >= 3 * TOT) return;
  int mat = i / TOT;
  int r   = i - mat * TOT;
  int d   = r % DHH;
  int bt  = r / DHH;
  const float* Wm  = (mat == 0) ? Wq : (mat == 1) ? Wk : Wv;
  const float* bm  = (mat == 0) ? bq : (mat == 1) ? bk : bv;
  const float* row = seq + (size_t)bt * DHH;
  const float* wr  = Wm + d * DHH;
  float a = bm[d];
#pragma unroll
  for (int j = 0; j < DHH; ++j) a += row[j] * wr[j];
  ((mat == 0) ? q : (mat == 1) ? k : v)[r] = a;
}

// ------------------------------------------ masked softmax attention + out-proj
__global__ void k_attn(const float* __restrict__ q, const float* __restrict__ k,
                       const float* __restrict__ v,
                       const float* __restrict__ Wo, const float* __restrict__ bo,
                       float* __restrict__ ctx) {
  int b   = blockIdx.x;
  int tid = threadIdx.x;   // 256
  __shared__ float s_q[DHH];
  __shared__ float s_p[TTT];
  __shared__ float s_cr[DHH];
  const float* qb = q + (size_t)b * TTT * DHH;
  const float* kb = k + (size_t)b * TTT * DHH;
  const float* vb = v + (size_t)b * TTT * DHH;
  const float  rs = rsqrtf((float)DHH);

  for (int t = 0; t < TTT; ++t) {
    if (tid < DHH) s_q[tid] = qb[t * DHH + tid];
    __syncthreads();
    if (tid < TTT) {
      const float* kr = kb + tid * DHH;
      float sc = 0.0f;
#pragma unroll
      for (int j = 0; j < DHH; ++j) sc += s_q[j] * kr[j];
      sc *= rs;
      bool m = (t == 0) ? (tid < 2) : (t == 1) ? (tid != 1) : (tid == 1);
      s_p[tid] = m ? sc : -1e9f;
    }
    __syncthreads();
    float mx = -1e30f;
    for (int s2 = 0; s2 < TTT; ++s2) mx = fmaxf(mx, s_p[s2]);
    __syncthreads();
    if (tid < TTT) s_p[tid] = __expf(s_p[tid] - mx);
    __syncthreads();
    float sum = 0.0f;
    for (int s2 = 0; s2 < TTT; ++s2) sum += s_p[s2];
    float inv = 1.0f / sum;
    if (tid < DHH) {
      float a = 0.0f;
      for (int s2 = 0; s2 < TTT; ++s2) a += s_p[s2] * vb[s2 * DHH + tid];
      s_cr[tid] = a * inv;
    }
    __syncthreads();
    if (tid < DHH) {
      const float* wr = Wo + tid * DHH;
      float o = bo[tid];
#pragma unroll
      for (int j = 0; j < DHH; ++j) o += s_cr[j] * wr[j];
      ctx[((size_t)b * TTT + t) * DHH + tid] = o;
    }
    __syncthreads();
  }
}

// ------------------------------------------------- fc_in: y = A @ W1^T + b1
// A[128,9800], W1[4340,9800]. Wave = one 16-wide N tile x FOUR 16-row M tiles
// (B fragment reused 4x -> W1 HBM traffic ~2x its size). grid(34, 2), 8
// waves/block. Main K loop is branch-free float4 loads; 8-element K tail is a
// single clamped+selected iteration. M geometry is exact (2*4*16 = 128).
__global__ void k_fc1_wmma(const float* __restrict__ A, const float* __restrict__ W1,
                           const float* __restrict__ b1, float* __restrict__ y,
                           int K, int Nn) {
  int lane = threadIdx.x & 31;
  int nt   = blockIdx.x * (blockDim.x >> 5) + (threadIdx.x >> 5);
  int NT   = (Nn + 15) >> 4;
  if (nt >= NT) return;
  int mg   = blockIdx.y * 4;            // first of 4 M-tiles
  int ncol = nt * 16 + (lane & 15);
  int ncl  = (ncol < Nn) ? ncol : (Nn - 1);   // clamp: loads always legal
  int ka   = (lane >> 4) * 8;
  int kb   = (lane >> 4) * 16;

  const float* arow[4];
#pragma unroll
  for (int mi = 0; mi < 4; ++mi)
    arow[mi] = A + (size_t)((mg + mi) * 16 + (lane & 15)) * K;
  const float* wrow = W1 + (size_t)ncl * K;

  v8f acc[4];
#pragma unroll
  for (int mi = 0; mi < 4; ++mi) acc[mi] = {};

  int kfull = (K / 32) * 32;   // 9792
  for (int k0 = 0; k0 < kfull; k0 += 32) {
    v16h bf;
    {
      float4 b0 = *(const float4*)(wrow + k0 + ka);
      float4 b1v = *(const float4*)(wrow + k0 + ka + 4);
      float4 b2 = *(const float4*)(wrow + k0 + 16 + ka);
      float4 b3 = *(const float4*)(wrow + k0 + 16 + ka + 4);
      pack8(bf, 0, b0, b1v);
      pack8(bf, 8, b2, b3);
    }
#pragma unroll
    for (int mi = 0; mi < 4; ++mi) {
      float4 a0 = *(const float4*)(arow[mi] + k0 + ka);
      float4 a1 = *(const float4*)(arow[mi] + k0 + ka + 4);
      float4 a2 = *(const float4*)(arow[mi] + k0 + 16 + ka);
      float4 a3 = *(const float4*)(arow[mi] + k0 + 16 + ka + 4);
      v16h a;
      pack8(a, 0, a0, a1);
      pack8(a, 8, a2, a3);
      acc[mi] = wmma16(a, bf, acc[mi]);
    }
  }
  // K tail (8 elements): unconditional clamped loads + select to zero
  {
    int k0 = kfull;
    v16h bf;
#pragma unroll
    for (int e = 0; e < 16; ++e) {
      int kk = k0 + ((e < 8) ? (ka + e) : (16 + ka + e - 8));
      float val = wrow[(kk < K) ? kk : (K - 1)];
      bf[e] = (_Float16)((kk < K) ? val : 0.0f);
    }
#pragma unroll
    for (int mi = 0; mi < 4; ++mi) {
      v16h a;
#pragma unroll
      for (int v = 0; v < 8; ++v) {
#pragma unroll
        for (int p = 0; p < 2; ++p) {
          int kk = k0 + ((v < 4) ? (ka + 2 * v + p) : (16 + ka + 2 * (v - 4) + p));
          float val = arow[mi][(kk < K) ? kk : (K - 1)];
          a[2 * v + p] = (_Float16)((kk < K) ? val : 0.0f);
        }
      }
      acc[mi] = wmma16(a, bf, acc[mi]);
    }
  }

  if (ncol < Nn) {
    float bias = b1[ncol];
    int mb = 8 * (lane >> 4);
#pragma unroll
    for (int mi = 0; mi < 4; ++mi) {
#pragma unroll
      for (int g = 0; g < 8; ++g)
        y[(size_t)((mg + mi) * 16 + mb + g) * Nn + ncol] = acc[mi][g] + bias;
    }
  }
}

// ------------------------------------------------------------- BN statistics
__global__ void k_bnstats(const float* __restrict__ y, float* __restrict__ mu,
                          float* __restrict__ var, int C) {
  int c = blockIdx.x * blockDim.x + threadIdx.x;
  if (c >= C) return;
  float m = 0.0f;
  for (int b = 0; b < BB; ++b) m += y[(size_t)b * C + c];
  m *= (1.0f / BB);
  float vv = 0.0f;
  for (int b = 0; b < BB; ++b) { float d = y[(size_t)b * C + c] - m; vv += d * d; }
  mu[c]  = m;
  var[c] = vv * (1.0f / BB);
}

// -------------------------------------------- BN apply + Mish + fc_out (dot)
__global__ void k_fc2(const float* __restrict__ y, const float* __restrict__ mu,
                      const float* __restrict__ var, const float* __restrict__ gamma,
                      const float* __restrict__ beta, const float* __restrict__ W2,
                      const float* __restrict__ b2, float* __restrict__ out, int C) {
  int b   = blockIdx.x;
  int tid = threadIdx.x;   // 256
  float acc = 0.0f;
  for (int c = tid; c < C; c += 256) {
    float v  = (y[(size_t)b * C + c] - mu[c]) * rsqrtf(var[c] + 1e-5f) * gamma[c] + beta[c];
    float sp = (v > 20.0f) ? v : log1pf(__expf(v));
    acc += v * tanhf(sp) * W2[c];
  }
  __shared__ float red[256];
  red[tid] = acc;
  __syncthreads();
  for (int s = 128; s > 0; s >>= 1) {
    if (tid < s) red[tid] += red[tid + s];
    __syncthreads();
  }
  if (tid == 0) out[b] = red[0] + b2[0];
}

// ---------------------------------------------------------------------------
extern "C" void kernel_launch(void* const* d_in, const int* in_sizes, int n_in,
                              void* d_out, int out_size, void* d_ws, size_t ws_size,
                              hipStream_t stream) {
  (void)n_in; (void)out_size; (void)ws_size;

  const float* x     = (const float*)d_in[0];
  const int*   ei    = (const int*)d_in[1];
  const int*   split = (const int*)d_in[2];
  const int*   batch = (const int*)d_in[3];
  const float* Wp    = (const float*)d_in[4];
  const float* bp    = (const float*)d_in[5];
  const float* Wl    = (const float*)d_in[6];
  const float* bl    = (const float*)d_in[7];
  const float* wif   = (const float*)d_in[8];
  const float* whf   = (const float*)d_in[9];
  const float* bif   = (const float*)d_in[10];
  const float* bhf   = (const float*)d_in[11];
  const float* wib   = (const float*)d_in[12];
  const float* whb   = (const float*)d_in[13];
  const float* bib   = (const float*)d_in[14];
  const float* bhb   = (const float*)d_in[15];
  const float* Wq    = (const float*)d_in[16];
  const float* bq    = (const float*)d_in[17];
  const float* Wk    = (const float*)d_in[18];
  const float* bk    = (const float*)d_in[19];
  const float* Wv    = (const float*)d_in[20];
  const float* bv    = (const float*)d_in[21];
  const float* Wo    = (const float*)d_in[22];
  const float* bo    = (const float*)d_in[23];
  const float* W1    = (const float*)d_in[24];
  const float* b1    = (const float*)d_in[25];
  const float* gamma = (const float*)d_in[26];
  const float* beta  = (const float*)d_in[27];
  const float* W2    = (const float*)d_in[28];
  const float* b2    = (const float*)d_in[29];
  float* out = (float*)d_out;

  const int N  = in_sizes[0] / DD;
  const int E  = in_sizes[1] / 2;
  const int C1 = 4340;           // fc_in cols  (W1 rows)
  const int K1 = TTT * DHH;      // 9800
  const size_t NW = (size_t)N * DD;

  const int* src = ei;
  const int* dst = ei + E;

  // ---- workspace carve-up (floats) ----
  float* Wf = (float*)d_ws;
  size_t o = 0;
  float* hbuf  = Wf + o; o += NW;
  float* tbuf  = Wf + o; o += NW;
  float* agg   = Wf + o; o += NW;
  float* dinvp = Wf + o; o += N;
  float* dinvl = Wf + o; o += N;
  float* pooled = Wf + o; o += 2 * BB * DD;       // [lig | pro]
  float* lstm  = Wf + o; o += (size_t)BB * TTT * DHH;
  float* qb    = Wf + o; o += (size_t)BB * TTT * DHH;
  float* kb    = Wf + o; o += (size_t)BB * TTT * DHH;
  float* vb    = Wf + o; o += (size_t)BB * TTT * DHH;
  float* ctxb  = Wf + o; o += (size_t)BB * TTT * DHH;
  float* ybuf  = Wf + o; o += (size_t)BB * C1;
  float* mu    = Wf + o; o += C1;
  float* var   = Wf + o; o += C1;

  const int TB = 256;
  const int ZB = 4096;

  // ---- degrees / symmetric normalization ----
  k_fill0<<<ZB, TB, 0, stream>>>(dinvp, (size_t)2 * N);  // dinvp,dinvl contiguous
  k_deg<<<(E + TB - 1) / TB, TB, 0, stream>>>(src, dst, split, dinvp, dinvl, E);
  k_dinv<<<(N + TB - 1) / TB, TB, 0, stream>>>(dinvp, dinvl, N);

  const int xform_blocks = (N + 127) / 128;           // 8 waves/block, 128 nodes/block
  const int upd_blocks   = (int)((NW + TB - 1) / TB);
  const int edge_blocks  = (E + TB - 1) / TB;

  // ---- protein branch (edge weight: split[src]==1) ----
  for (int i = 0; i < 5; ++i) {
    const float* hin = (i == 0) ? x : hbuf;
    k_xform_wmma<<<xform_blocks, TB, 0, stream>>>(hin, Wp + i * DD * DD, tbuf, N);
    k_fill0<<<ZB, TB, 0, stream>>>(agg, NW);
    k_edge_agg<<<edge_blocks, TB, 0, stream>>>(src, dst, split, 1, dinvp, tbuf, agg, E);
    k_node_update<<<upd_blocks, TB, 0, stream>>>(tbuf, agg, dinvp, bp + i * DD, hbuf, (int)NW);
  }
  k_fill0<<<8, TB, 0, stream>>>(pooled + BB * DD, (size_t)BB * DD);
  k_pool<<<upd_blocks, TB, 0, stream>>>(hbuf, batch, pooled + BB * DD, (int)NW);

  // ---- ligand branch (edge weight: split[src]==0) ----
  for (int i = 0; i < 5; ++i) {
    const float* hin = (i == 0) ? x : hbuf;
    k_xform_wmma<<<xform_blocks, TB, 0, stream>>>(hin, Wl + i * DD * DD, tbuf, N);
    k_fill0<<<ZB, TB, 0, stream>>>(agg, NW);
    k_edge_agg<<<edge_blocks, TB, 0, stream>>>(src, dst, split, 0, dinvl, tbuf, agg, E);
    k_node_update<<<upd_blocks, TB, 0, stream>>>(tbuf, agg, dinvl, bl + i * DD, hbuf, (int)NW);
  }
  k_fill0<<<8, TB, 0, stream>>>(pooled, (size_t)BB * DD);
  k_pool<<<upd_blocks, TB, 0, stream>>>(hbuf, batch, pooled, (int)NW);

  // ---- BiLSTM over padded T=140 sequence ----
  k_lstm<<<2 * BB, 160, 0, stream>>>(pooled, pooled + BB * DD,
                                     wif, whf, bif, bhf, wib, whb, bib, bhb, lstm);

  // ---- single-head masked self-attention ----
  {
    int tot = 3 * BB * TTT * DHH;
    k_qkv<<<(tot + TB - 1) / TB, TB, 0, stream>>>(lstm, Wq, bq, Wk, bk, Wv, bv, qb, kb, vb);
    k_attn<<<BB, TB, 0, stream>>>(qb, kb, vb, Wo, bo, ctxb);
  }

  // ---- fc_in GEMM (WMMA) + BN(train) + Mish + fc_out ----
  {
    dim3 grid((((C1 + 15) / 16) + 7) / 8, 2);   // 272 N-tiles, 2 groups of 4 M-tiles
    k_fc1_wmma<<<grid, TB, 0, stream>>>(ctxb, W1, b1, ybuf, K1, C1);
  }
  k_bnstats<<<(C1 + TB - 1) / TB, TB, 0, stream>>>(ybuf, mu, var, C1);
  k_fc2<<<BB, TB, 0, stream>>>(ybuf, mu, var, gamma, beta, W2, b2, out, C1);
}
